// PhyloGNN_56221121905069
// MI455X (gfx1250) — matrix operand order repaired and checked by reading, not compiled, not run on hardware
//
#include <hip/hip_runtime.h>

#define F0    104   // HID + EMB + DIST
#define HID   48
#define EMB   24
#define DIST  32
#define VOCAB 5000

typedef __attribute__((ext_vector_type(16))) __bf16 v16bf;
typedef __attribute__((ext_vector_type(8)))  __bf16 v8bf;
typedef __attribute__((ext_vector_type(8)))  float  v8f;
typedef __attribute__((ext_vector_type(4)))  unsigned int u32x4;
typedef __attribute__((ext_vector_type(8)))  int i32x8;
typedef __attribute__((ext_vector_type(4)))  int i32x4;

__device__ __forceinline__ float relu_f(float x) { return fmaxf(x, 0.0f); }

// ---------------------------------------------------------------------------
// TDM: 1-row tensor load of `nelem` f32 from global -> LDS (D# per ISA ch.8).
//   group0: count=1, lds_addr, global_addr[56:0], type=2
//   group1: data_size=4B, tensor_dim0=nelem, tensor_dim1=1, tile_dim0=nelem,
//           tile_dim1=1, stride0=nelem
// ---------------------------------------------------------------------------
#if __has_builtin(__builtin_amdgcn_tensor_load_to_lds)
#define PHYLO_HAS_TDM 1
__device__ __forceinline__ void phylo_tdm_load_f32(unsigned lds_off,
                                                   const float* gptr,
                                                   unsigned nelem)
{
    unsigned long long ga = (unsigned long long)(uintptr_t)gptr;
    u32x4 g0;
    g0.x = 1u;                                               // count=1
    g0.y = lds_off;                                          // lds_addr
    g0.z = (unsigned)(ga & 0xFFFFFFFFu);                     // global_addr[31:0]
    g0.w = (unsigned)((ga >> 32) & 0x01FFFFFFu) | (2u << 30);// addr[56:32] | type=2
    i32x8 g1;
    g1[0] = (2 << 16);                                       // data_size = 4 bytes
    g1[1] = (int)((nelem & 0xFFFFu) << 16);                  // tensor_dim0[15:0]
    g1[2] = (int)((nelem >> 16) | (1u << 16));               // dim0[31:16] | tensor_dim1=1
    g1[3] = (int)(nelem << 16);                              // tensor_dim1 hi=0 | tile_dim0
    g1[4] = 1;                                               // tile_dim1=1, tile_dim2=0
    g1[5] = (int)nelem;                                      // tensor_dim0_stride[31:0]
    g1[6] = 0;
    g1[7] = 0;
    i32x4 z4 = {0, 0, 0, 0};
#if defined(__clang_major__) && __clang_major__ >= 23
    i32x8 z8 = {0, 0, 0, 0, 0, 0, 0, 0};
    __builtin_amdgcn_tensor_load_to_lds(g0, g1, z4, z4, z8, 0);
#else
    __builtin_amdgcn_tensor_load_to_lds(g0, g1, z4, z4, 0);
#endif
}
#else
#define PHYLO_HAS_TDM 0
#endif

// ---------------------------------------------------------------------------
// 1) Node feature construction (coalesced, one thread per output element)
// ---------------------------------------------------------------------------
__global__ __launch_bounds__(256)
void phylo_node_feat_kernel(const float* __restrict__ ab,
                            const int*   __restrict__ taxo,
                            const float* __restrict__ dist,
                            const float* __restrict__ emb,
                            const float* __restrict__ ap_w, const float* __restrict__ ap_b,
                            const float* __restrict__ dp_w, const float* __restrict__ dp_b,
                            float* __restrict__ h0, int n)
{
    long idx = (long)blockIdx.x * blockDim.x + threadIdx.x;
    long total = (long)n * F0;
    if (idx >= total) return;
    int i = (int)(idx / F0);
    int f = (int)(idx % F0);
    float v;
    if (f < HID) {
        v = relu_f(ab[i] * ap_w[f] + ap_b[f]);
    } else if (f < HID + EMB) {
        int t = taxo[i];
        t = min(max(t, 0), VOCAB - 1);
        v = emb[(long)t * EMB + (f - HID)];
    } else {
        int g = f - (HID + EMB);
        v = relu_f(dist[i] * dp_w[g] + dp_b[g]);
    }
    h0[idx] = v;
}

// ---------------------------------------------------------------------------
// 2) Edge scatter-add: agg[dst] += h[src].  float4 gather + 4 f32 atomics.
// ---------------------------------------------------------------------------
__global__ __launch_bounds__(256)
void phylo_scatter_add_kernel(const float* __restrict__ h,
                              float* __restrict__ agg,
                              const int* __restrict__ src,
                              const int* __restrict__ dst,
                              int nE, int F, int FV /* = F/4 */)
{
    long idx = (long)blockIdx.x * blockDim.x + threadIdx.x;
    long total = (long)nE * FV;
    if (idx >= total) return;
    int e = (int)(idx / FV);
    int c = (int)(idx % FV) * 4;
    int s = src[e];
    int d = dst[e];
    const float4 v = *(const float4*)(h + (long)s * F + c);
    float* p = agg + (long)d * F + c;
    atomicAdd(p + 0, v.x);
    atomicAdd(p + 1, v.y);
    atomicAdd(p + 2, v.z);
    atomicAdd(p + 3, v.w);
}

// ---------------------------------------------------------------------------
// 3) GIN MLP via bf16 WMMA:  out = relu( relu(z@W1 + b1) @ W2 + b2 ),
//    z = (1+eps)*h + agg.  Templated on KP (padded K: 128 for F=104, 64 for 48).
//    Weights TDM-loaded (f32) into LDS, repacked into per-lane fragment order
//    so every B-fragment is one contiguous 32B LDS load (2x ds_load_b128).
// ---------------------------------------------------------------------------
template <int F, int KP>
__global__ __launch_bounds__(128)
void phylo_gin_mlp_kernel(const float* __restrict__ h,
                          const float* __restrict__ agg,
                          const float* __restrict__ w1,   // [F, 48] row-major
                          const float* __restrict__ b1,   // [48]
                          const float* __restrict__ w2,   // [48, 48]
                          const float* __restrict__ b2,   // [48]
                          const float* __restrict__ eps_p,
                          float* __restrict__ out,        // [n, 48]
                          int n)
{
    constexpr int NK1 = KP / 32;        // k-steps layer 1
    constexpr int NK2 = 2;              // k-steps layer 2 (K=64 padded)

    __shared__ __align__(16) float  w1raw[F0 * HID];        // TDM landing zone
    __shared__ __align__(16) float  w2raw[HID * HID];
    __shared__ __align__(32) __bf16 w1f[3][NK1][32][16];    // fragment-ordered
    __shared__ __align__(32) __bf16 w2f[3][NK2][32][16];
    __shared__ __align__(32) __bf16 zs[4][16][KP];          // per-wave z tile
    __shared__ __align__(32) __bf16 mid[4][16][64];         // per-wave intermediate
    __shared__ float b1s[HID], b2s[HID];

    const int tid  = threadIdx.x;
    const int wave = tid >> 5;
    const int lane = tid & 31;

    // ---- stage raw f32 weights into LDS ----
#if PHYLO_HAS_TDM
    if (wave == 0) {
        phylo_tdm_load_f32((unsigned)(uintptr_t)&w1raw[0], w1, F * HID);
        phylo_tdm_load_f32((unsigned)(uintptr_t)&w2raw[0], w2, HID * HID);
        __builtin_amdgcn_s_wait_tensorcnt(0);
    }
#else
    for (int i = tid; i < F * HID; i += 128) w1raw[i] = w1[i];
    for (int i = tid; i < HID * HID; i += 128) w2raw[i] = w2[i];
#endif
    if (tid < HID) { b1s[tid] = b1[tid]; b2s[tid] = b2[tid]; }
    const float epsp1 = 1.0f + eps_p[0];
    __syncthreads();

    // ---- repack weights into per-lane WMMA B-fragment order (bf16) ----
    // B element j of lane L: K = ks*32 + (L>=16 ? 8 : 0) + j + (j>=8 ? 8 : 0), N = nt*16 + (L&15)
    for (int idx = tid; idx < 3 * NK1 * 32 * 16; idx += 128) {
        int j  = idx & 15;
        int ln = (idx >> 4) & 31;
        int ks = (idx >> 9) % NK1;
        int nt = idx / (NK1 * 512);
        int k  = ks * 32 + ((ln & 16) ? 8 : 0) + j + ((j >= 8) ? 8 : 0);
        int c  = nt * 16 + (ln & 15);
        w1f[nt][ks][ln][j] = (__bf16)((k < F) ? w1raw[k * HID + c] : 0.0f);
    }
    for (int idx = tid; idx < 3 * NK2 * 32 * 16; idx += 128) {
        int j  = idx & 15;
        int ln = (idx >> 4) & 31;
        int ks = (idx >> 9) % NK2;
        int nt = idx / (NK2 * 512);
        int k  = ks * 32 + ((ln & 16) ? 8 : 0) + j + ((j >= 8) ? 8 : 0);
        int c  = nt * 16 + (ln & 15);
        w2f[nt][ks][ln][j] = (__bf16)((k < HID) ? w2raw[k * HID + c] : 0.0f);
    }

    const int baseM   = (blockIdx.x * 4 + wave) * 16;
    const bool active = (baseM < n);    // wave-uniform -> EXEC all-ones inside

    // ---- stage z tile (bf16, zero-padded K) + zero mid padding ----
    if (active) {
        for (int idx = lane; idx < 16 * KP; idx += 32) {
            int r = idx / KP, c = idx % KP;
            float v = 0.0f;
            if (c < F) {
                long off = (long)(baseM + r) * F + c;
                v = epsp1 * h[off] + agg[off];
            }
            zs[wave][r][c] = (__bf16)v;
        }
        for (int idx = lane; idx < 16 * 64; idx += 32)
            mid[wave][idx >> 6][idx & 63] = (__bf16)0.0f;
    }
    __syncthreads();

    const int mrow  = lane & 15;
    const int koffA = (lane & 16) ? 8 : 0;   // A-fragment K bias (16-bit layout)
    const int ncol  = lane & 15;

    // ---- layer 1: z[16xKP] @ W1[KPx48] ----
    if (active) {
#pragma unroll
        for (int nt = 0; nt < 3; ++nt) {
            v8f acc = {};
#pragma unroll
            for (int ks = 0; ks < NK1; ++ks) {
                const int kb = ks * 32 + koffA;
                v8bf lo = *(const v8bf*)&zs[wave][mrow][kb];
                v8bf hi = *(const v8bf*)&zs[wave][mrow][kb + 16];
                v16bf A = __builtin_shufflevector(lo, hi,
                            0, 1, 2, 3, 4, 5, 6, 7, 8, 9, 10, 11, 12, 13, 14, 15);
                v16bf B = *(const v16bf*)&w1f[nt][ks][lane][0];
                acc = __builtin_amdgcn_wmma_f32_16x16x32_bf16(
                        false, A, false, B, (short)0, acc, false, false);
            }
            const int nn = nt * 16 + ncol;
            const float bias = b1s[nn];
#pragma unroll
            for (int g = 0; g < 8; ++g) {
                int m = g + ((lane & 16) ? 8 : 0);
                mid[wave][m][nn] = (__bf16)relu_f(acc[g] + bias);
            }
        }
    }
    __syncthreads();

    // ---- layer 2: mid[16x64] @ W2[64x48] ----
    if (active) {
#pragma unroll
        for (int nt = 0; nt < 3; ++nt) {
            v8f acc = {};
#pragma unroll
            for (int ks = 0; ks < NK2; ++ks) {
                const int kb = ks * 32 + koffA;
                v8bf lo = *(const v8bf*)&mid[wave][mrow][kb];
                v8bf hi = *(const v8bf*)&mid[wave][mrow][kb + 16];
                v16bf A = __builtin_shufflevector(lo, hi,
                            0, 1, 2, 3, 4, 5, 6, 7, 8, 9, 10, 11, 12, 13, 14, 15);
                v16bf B = *(const v16bf*)&w2f[nt][ks][lane][0];
                acc = __builtin_amdgcn_wmma_f32_16x16x32_bf16(
                        false, A, false, B, (short)0, acc, false, false);
            }
            const int nn = nt * 16 + ncol;
            const float bias = b2s[nn];
#pragma unroll
            for (int g = 0; g < 8; ++g) {
                int m = g + ((lane & 16) ? 8 : 0);
                out[(long)(baseM + m) * HID + nn] = relu_f(acc[g] + bias);
            }
        }
    }
}

// ---------------------------------------------------------------------------
// 4) Global mean pool (atomics into [B,48] sums + [B] counts)
// ---------------------------------------------------------------------------
__global__ __launch_bounds__(256)
void phylo_pool_kernel(const float* __restrict__ h,
                       const int* __restrict__ batch,
                       float* __restrict__ sums, float* __restrict__ cnts, int n)
{
    long idx = (long)blockIdx.x * blockDim.x + threadIdx.x;
    long total = (long)n * HID;
    if (idx >= total) return;
    int i = (int)(idx / HID);
    int f = (int)(idx % HID);
    int b = batch[i];
    atomicAdd(&sums[(long)b * HID + f], h[idx]);
    if (f == 0) atomicAdd(&cnts[b], 1.0f);
}

// ---------------------------------------------------------------------------
// 5) Head MLP: [B,52] @ [52,64] -> relu -> [64,1]  (B=256, trivial, scalar)
// ---------------------------------------------------------------------------
__global__ __launch_bounds__(256)
void phylo_head_kernel(const float* __restrict__ sums, const float* __restrict__ cnts,
                       const float* __restrict__ cov,
                       const float* __restrict__ w1, const float* __restrict__ b1,
                       const float* __restrict__ w2, const float* __restrict__ b2,
                       float* __restrict__ out, int nB)
{
    int g = blockIdx.x * blockDim.x + threadIdx.x;
    if (g >= nB) return;
    float comb[HID + 4];
    float c = fmaxf(cnts[g], 1.0f);
    for (int k = 0; k < HID; ++k) comb[k] = sums[(long)g * HID + k] / c;
    for (int k = 0; k < 4;   ++k) comb[HID + k] = cov[(long)g * 4 + k];
    float o = b2[0];
    for (int j = 0; j < 64; ++j) {
        float hj = b1[j];
        for (int k = 0; k < HID + 4; ++k) hj += comb[k] * w1[(long)k * 64 + j];
        o += relu_f(hj) * w2[j];
    }
    out[g] = o;
}

// ---------------------------------------------------------------------------
extern "C" void kernel_launch(void* const* d_in, const int* in_sizes, int n_in,
                              void* d_out, int out_size, void* d_ws, size_t ws_size,
                              hipStream_t stream)
{
    const float* abundance = (const float*)d_in[0];
    const int*   taxo      = (const int*)  d_in[1];
    const float* distance  = (const float*)d_in[2];
    const int*   edges     = (const int*)  d_in[3];
    const int*   batch     = (const int*)  d_in[4];
    const float* cov       = (const float*)d_in[5];
    const float* emb       = (const float*)d_in[6];
    const float* ap_w  = (const float*)d_in[7];
    const float* ap_b  = (const float*)d_in[8];
    const float* dp_w  = (const float*)d_in[9];
    const float* dp_b  = (const float*)d_in[10];
    const float* c0_w1 = (const float*)d_in[11];
    const float* c0_b1 = (const float*)d_in[12];
    const float* c0_w2 = (const float*)d_in[13];
    const float* c0_b2 = (const float*)d_in[14];
    const float* eps0  = (const float*)d_in[15];
    const float* c1_w1 = (const float*)d_in[16];
    const float* c1_b1 = (const float*)d_in[17];
    const float* c1_w2 = (const float*)d_in[18];
    const float* c1_b2 = (const float*)d_in[19];
    const float* eps1  = (const float*)d_in[20];
    const float* h_w1  = (const float*)d_in[21];
    const float* h_b1  = (const float*)d_in[22];
    const float* h_w2  = (const float*)d_in[23];
    const float* h_b2  = (const float*)d_in[24];

    const int n  = in_sizes[0];
    const int nE = in_sizes[3] / 2;
    const int nB = in_sizes[5] / 4;
    const int* src = edges;
    const int* dst = edges + nE;

    // workspace (floats): bufA[n*104] | bufB[n*104] | bufC[n*48] | psum[B*48] | pcnt[B]
    float* ws   = (float*)d_ws;
    float* bufA = ws;
    float* bufB = ws + (size_t)n * F0;
    float* bufC = ws + (size_t)n * F0 * 2;
    float* psum = bufC + (size_t)n * HID;
    float* pcnt = psum + (size_t)nB * HID;

    const int mlpBlocks = (((n + 15) / 16) + 3) / 4;

    hipMemsetAsync(bufB, 0, (size_t)n * F0 * sizeof(float), stream);
    hipMemsetAsync(psum, 0, (size_t)nB * (HID + 1) * sizeof(float), stream);

    {   // h0
        long total = (long)n * F0;
        phylo_node_feat_kernel<<<(int)((total + 255) / 256), 256, 0, stream>>>(
            abundance, taxo, distance, emb, ap_w, ap_b, dp_w, dp_b, bufA, n);
    }
    {   // agg0 = scatter(h0)
        long total = (long)nE * (F0 / 4);
        phylo_scatter_add_kernel<<<(int)((total + 255) / 256), 256, 0, stream>>>(
            bufA, bufB, src, dst, nE, F0, F0 / 4);
    }
    // h1 = GIN-MLP0 (WMMA, KP=128)
    phylo_gin_mlp_kernel<F0, 128><<<mlpBlocks, 128, 0, stream>>>(
        bufA, bufB, c0_w1, c0_b1, c0_w2, c0_b2, eps0, bufC, n);

    hipMemsetAsync(bufB, 0, (size_t)n * HID * sizeof(float), stream);
    {   // agg1 = scatter(h1)
        long total = (long)nE * (HID / 4);
        phylo_scatter_add_kernel<<<(int)((total + 255) / 256), 256, 0, stream>>>(
            bufC, bufB, src, dst, nE, HID, HID / 4);
    }
    // h2 = GIN-MLP1 (WMMA, KP=64), overwrite bufA (h0 dead)
    phylo_gin_mlp_kernel<HID, 64><<<mlpBlocks, 128, 0, stream>>>(
        bufC, bufB, c1_w1, c1_b1, c1_w2, c1_b2, eps1, bufA, n);

    {   // mean pool
        long total = (long)n * HID;
        phylo_pool_kernel<<<(int)((total + 255) / 256), 256, 0, stream>>>(
            bufA, batch, psum, pcnt, n);
    }
    // head
    phylo_head_kernel<<<(nB + 255) / 256, 256, 0, stream>>>(
        psum, pcnt, cov, h_w1, h_b1, h_w2, h_b2, (float*)d_out, nB);
}
